// AttentionLayer_63024350101725
// MI455X (gfx1250) — compile-verified, matrix operand order checked by reading, not run on hardware
//
#include <hip/hip_runtime.h>
#include <cstdint>

// ---------------------------------------------------------------------------
// Transformer block (pre-norm MHA + FFN) for MI455X (gfx1250, wave32, WMMA).
// GEMMs: v_wmma_f32_16x16x32_bf16, LDS tiles staged by the Tensor Data Mover
// (tensor_load_to_lds, TENSORcnt-tracked, double-buffered).  LN / softmax /
// residuals in f32.  Attention is causal flash-style (online softmax).
// ---------------------------------------------------------------------------

#define C_DIM  4096
#define E_DIM  1024
#define NH_    16
#define HS_    64
#define D_DIM  1024
#define FF_DIM 4096

typedef __attribute__((ext_vector_type(16))) __bf16 v16bf;
typedef __attribute__((ext_vector_type(8)))  float  v8f;
typedef __attribute__((ext_vector_type(4)))  unsigned int u32x4;
typedef __attribute__((ext_vector_type(8)))  int  i32x8;
typedef __attribute__((ext_vector_type(4)))  int  i32x4;

union BFrag { v16bf v; uint4 q[2]; unsigned short us[16]; };
union FAcc  { v8f   v; float f[8]; };

__device__ __forceinline__ unsigned short f32_to_bf16(float f) {
  unsigned int u = __float_as_uint(f);
  unsigned int r = u + 0x7FFFu + ((u >> 16) & 1u);   // round-to-nearest-even
  return (unsigned short)(r >> 16);
}

__device__ __forceinline__ v8f wmma_bf16(BFrag a, BFrag b, v8f c) {
  return __builtin_amdgcn_wmma_f32_16x16x32_bf16(
      /*neg_a=*/false, a.v, /*neg_b=*/false, b.v,
      /*c_mod=*/(short)0, c, /*reuse_a=*/false, /*reuse_b=*/false);
}

// ---------------------------------------------------------------------------
// TDM: DMA a 2D tile (rows x k_dw dwords) from global into LDS, with the
// hardware inserting 4 pad dwords after every 32 data dwords so the LDS tile
// keeps a (BK/2 + 4)-dword pitch (bank-conflict padding).  D# per ISA ch.8:
//   group0: count=1 | lds_addr | global tile addr (57b) | type=2
//   group1: data_size=4B, pad_enable, pad_interval=32dw(code 4),
//           pad_amount=4dw(code 3), tensor dims huge (tiles never OOB),
//           tile_dim0=k_dw, tile_dim1=rows, dim0 stride = pitch_dw
// ---------------------------------------------------------------------------
__device__ __forceinline__ void tdm_load_tile(
    const unsigned short* g, const void* lds, int rows, int k_dw,
    long long pitch_dw)
{
  const unsigned long long ga = (unsigned long long)(size_t)g;
  const unsigned lds_off = (unsigned)(size_t)lds;   // low 32b == LDS offset
  const u32x4 g0 = { 1u, lds_off,
                     (unsigned)(ga & 0xFFFFFFFFull),
                     (unsigned)((ga >> 32) & 0x01FFFFFFull) | (2u << 30) };
  const unsigned d0 = 1u << 20;     // tensor_dim0 (OOB never hit)
  const unsigned d1 = 1u << 20;     // tensor_dim1
  const unsigned long long s0 = (unsigned long long)pitch_dw;
  const i32x8 g1 = {
      (int)((2u << 16) | (1u << 20) | (4u << 22) | (3u << 25)),
      (int)((d0 & 0xFFFFu) << 16),
      (int)((d0 >> 16) | ((d1 & 0xFFFFu) << 16)),
      (int)((d1 >> 16) | ((unsigned)k_dw << 16)),
      (int)(unsigned)rows,
      (int)(unsigned)(s0 & 0xFFFFFFFFull),
      (int)(unsigned)((s0 >> 32) & 0xFFFFull),
      0 };
  const i32x4 z4 = { 0, 0, 0, 0 };
#if defined(__clang_major__) && (__clang_major__ >= 23)
  const i32x8 z8 = { 0, 0, 0, 0, 0, 0, 0, 0 };
  __builtin_amdgcn_tensor_load_to_lds(g0, g1, z4, z4, z8, 0);
#else
  __builtin_amdgcn_tensor_load_to_lds(g0, g1, z4, z4, 0);
#endif
}

// ---------------------------------------------------------------------------
// f32 [Kd][Nd] row-major  ->  bf16 [Nd][Kd] (N-major, for B fragments)
// ---------------------------------------------------------------------------
__global__ __launch_bounds__(256) void cvt_transpose_kernel(
    const float* __restrict__ in, unsigned short* __restrict__ out,
    int Kd, int Nd)
{
  __shared__ float tile[32][33];
  const long long base = (long long)blockIdx.z * Kd * Nd;
  in  += base;
  out += base;
  const int n0 = blockIdx.x * 32;
  const int k0 = blockIdx.y * 32;
  const int tx = threadIdx.x & 31;
  const int ty = threadIdx.x >> 5;
  for (int r = ty; r < 32; r += 8)
    tile[r][tx] = in[(long long)(k0 + r) * Nd + n0 + tx];
  __syncthreads();
  for (int r = ty; r < 32; r += 8)
    out[(long long)(n0 + r) * Kd + k0 + tx] = f32_to_bf16(tile[tx][r]);
}

// ---------------------------------------------------------------------------
// LayerNorm over last dim (1024) -> bf16.  One block (256 threads) per row.
// ---------------------------------------------------------------------------
__global__ __launch_bounds__(256) void ln_bf16_kernel(
    const float* __restrict__ x, const float* __restrict__ g,
    const float* __restrict__ b, unsigned short* __restrict__ out)
{
  const int row = blockIdx.x;
  const float* xr = x + (long long)row * E_DIM;
  const int i0 = threadIdx.x * 4;
  const float4 v = *(const float4*)&xr[i0];
  float s  = v.x + v.y + v.z + v.w;
  float s2 = v.x * v.x + v.y * v.y + v.z * v.z + v.w * v.w;
#pragma unroll
  for (int msk = 1; msk < 32; msk <<= 1) {
    s  += __shfl_xor(s,  msk, 32);
    s2 += __shfl_xor(s2, msk, 32);
  }
  __shared__ float ws[8], ws2[8];
  if ((threadIdx.x & 31) == 0) { ws[threadIdx.x >> 5] = s; ws2[threadIdx.x >> 5] = s2; }
  __syncthreads();
  float ts = 0.f, ts2 = 0.f;
#pragma unroll
  for (int i = 0; i < 8; i++) { ts += ws[i]; ts2 += ws2[i]; }
  const float mu  = ts * (1.0f / E_DIM);
  const float var = ts2 * (1.0f / E_DIM) - mu * mu;
  const float rs  = rsqrtf(var + 1e-5f);
  const float4 gg = *(const float4*)&g[i0];
  const float4 bb = *(const float4*)&b[i0];
  ushort4 ov;
  ov.x = f32_to_bf16((v.x - mu) * rs * gg.x + bb.x);
  ov.y = f32_to_bf16((v.y - mu) * rs * gg.y + bb.y);
  ov.z = f32_to_bf16((v.z - mu) * rs * gg.z + bb.z);
  ov.w = f32_to_bf16((v.w - mu) * rs * gg.w + bb.w);
  *(ushort4*)&out[(long long)row * E_DIM + i0] = ov;
}

// ---------------------------------------------------------------------------
// bf16 WMMA GEMM:  C[M,N] = A[M,K] * Bt[N,K]^T  (+bias, relu, residual)
// 8 waves (4 M x 2 N), BK=64, LDS double-buffered, tiles staged by the TDM
// (wave 0 issues tensor_load_to_lds for tile i+1 while all waves compute
// tile i; drained with s_wait_tensorcnt before the block barrier).
// ---------------------------------------------------------------------------
template<int BM, int BN, bool BIAS, bool RELU, bool RESID, bool OUT_BF16, bool OUT_TRANS>
__global__ __launch_bounds__(256, 2) void gemm_bf16_kernel(
    const unsigned short* __restrict__ A,
    const unsigned short* __restrict__ Bt,
    const float* __restrict__ bias,
    const float* __restrict__ resid,
    float* __restrict__ Of,
    unsigned short* __restrict__ Ob,
    int M, int N, int K,
    long long strideA, long long strideB, long long strideO)
{
  constexpr int BK = 64;
  __shared__ unsigned short Asm_[2][BM][BK + 8];
  __shared__ unsigned short Bsm_[2][BN][BK + 8];

  const int z = blockIdx.z;
  A  += (long long)z * strideA;
  Bt += (long long)z * strideB;

  const int wave  = threadIdx.x >> 5;
  const int lane  = threadIdx.x & 31;
  const int lrow  = lane & 15;
  const int lhalf = lane >> 4;
  constexpr int WM = BM / 4;
  constexpr int WN = BN / 2;
  constexpr int MT = WM / 16;
  constexpr int NT = WN / 16;
  const int wm = wave >> 1;
  const int wn = wave & 1;

  const int m0 = blockIdx.y * BM;
  const int n0 = blockIdx.x * BN;

  FAcc acc[MT][NT];
#pragma unroll
  for (int i = 0; i < MT; i++)
#pragma unroll
    for (int j = 0; j < NT; j++)
#pragma unroll
      for (int e = 0; e < 8; e++) acc[i][j].f[e] = 0.f;

  // prologue: DMA tile 0 into buffer 0
  if (wave == 0) {
    tdm_load_tile(&A[(long long)m0 * K], &Asm_[0][0][0], BM, BK / 2, K / 2);
    tdm_load_tile(&Bt[(long long)n0 * K], &Bsm_[0][0][0], BN, BK / 2, K / 2);
    __builtin_amdgcn_s_wait_tensorcnt(0);
  }
  __syncthreads();

  for (int k0 = 0; k0 < K; k0 += BK) {
    const int buf = (k0 / BK) & 1;
    // DMA next tile into the other buffer while this one is consumed
    if (wave == 0 && (k0 + BK) < K) {
      tdm_load_tile(&A[(long long)m0 * K + k0 + BK], &Asm_[buf ^ 1][0][0],
                    BM, BK / 2, K / 2);
      tdm_load_tile(&Bt[(long long)n0 * K + k0 + BK], &Bsm_[buf ^ 1][0][0],
                    BN, BK / 2, K / 2);
    }

#pragma unroll
    for (int kk = 0; kk < BK; kk += 32) {
      BFrag af[MT], bfr[NT];
#pragma unroll
      for (int i = 0; i < MT; i++) {
        const int r = wm * WM + i * 16 + lrow;
        af[i].q[0] = *(const uint4*)&Asm_[buf][r][kk + lhalf * 8];
        af[i].q[1] = *(const uint4*)&Asm_[buf][r][kk + lhalf * 8 + 16];
      }
#pragma unroll
      for (int j = 0; j < NT; j++) {
        const int r = wn * WN + j * 16 + lrow;
        bfr[j].q[0] = *(const uint4*)&Bsm_[buf][r][kk + lhalf * 8];
        bfr[j].q[1] = *(const uint4*)&Bsm_[buf][r][kk + lhalf * 8 + 16];
      }
#pragma unroll
      for (int i = 0; i < MT; i++)
#pragma unroll
        for (int j = 0; j < NT; j++)
          acc[i][j].v = wmma_bf16(af[i], bfr[j], acc[i][j].v);
    }

    if (wave == 0) __builtin_amdgcn_s_wait_tensorcnt(0);
    __syncthreads();
  }

  // ---- epilogue (C/D layout: lane -> col = l&15, row = vg + 8*(l>>4)) ----
#pragma unroll
  for (int i = 0; i < MT; i++) {
#pragma unroll
    for (int j = 0; j < NT; j++) {
      const int n = n0 + wn * WN + j * 16 + lrow;
#pragma unroll
      for (int e = 0; e < 8; e++) {
        const int m = m0 + wm * WM + i * 16 + e + 8 * lhalf;
        float v = acc[i][j].f[e];
        if (BIAS)  v += bias[n];
        if (RELU)  v  = fmaxf(v, 0.f);
        if (RESID) v += resid[(long long)m * N + n];
        if (OUT_BF16) {
          if (OUT_TRANS)
            Ob[(long long)z * strideO + (long long)n * M + m] = f32_to_bf16(v);
          else
            Ob[(long long)z * strideO + (long long)m * N + n] = f32_to_bf16(v);
        } else {
          Of[(long long)m * N + n] = v;
        }
      }
    }
  }
}

// ---------------------------------------------------------------------------
// Causal flash attention: one block per (head, 128 query rows); 8 waves,
// each wave owns 16 query rows; keys processed 32 at a time.
//   Q,K : [NH][C][HS] bf16    Vt : [NH][HS][C] bf16    O : [C][NH*HS] bf16
// Masking only on diagonal blocks (wave-uniform branch).
// ---------------------------------------------------------------------------
__global__ __launch_bounds__(256, 2) void flash_attn_kernel(
    const unsigned short* __restrict__ Q,
    const unsigned short* __restrict__ Kmat,
    const unsigned short* __restrict__ Vt,
    unsigned short* __restrict__ Obf,
    float scale)
{
  __shared__ unsigned short Ksm[32][HS_ + 8];
  __shared__ unsigned short Vsm[HS_][32 + 8];
  __shared__ unsigned short Psm[8][16][32 + 8];

  const int h     = blockIdx.y;
  const int qb    = blockIdx.x;
  const int wave  = threadIdx.x >> 5;
  const int lane  = threadIdx.x & 31;
  const int lrow  = lane & 15;
  const int lhalf = lane >> 4;
  const int m0    = qb * 128 + wave * 16;

  const unsigned short* Qh = Q    + (long long)h * C_DIM * HS_;
  const unsigned short* Kh = Kmat + (long long)h * C_DIM * HS_;
  const unsigned short* Vh = Vt   + (long long)h * HS_ * C_DIM;

  BFrag q0, q1;
  {
    const unsigned short* qr = Qh + (long long)(m0 + lrow) * HS_;
    q0.q[0] = *(const uint4*)&qr[lhalf * 8];
    q0.q[1] = *(const uint4*)&qr[lhalf * 8 + 16];
    q1.q[0] = *(const uint4*)&qr[32 + lhalf * 8];
    q1.q[1] = *(const uint4*)&qr[48 + lhalf * 8];
  }

  FAcc o[4];
#pragma unroll
  for (int t = 0; t < 4; t++)
#pragma unroll
    for (int e = 0; e < 8; e++) o[t].f[e] = 0.f;
  float mrow[8], lsum[8];
#pragma unroll
  for (int e = 0; e < 8; e++) { mrow[e] = -1e30f; lsum[e] = 0.f; }

  const int jend = qb * 128 + 128;
  for (int j0 = 0; j0 < jend; j0 += 32) {
    __syncthreads();
    for (int t = threadIdx.x; t < 32 * 8; t += 256) {
      const int r = t >> 3, c = (t & 7) << 3;
      *(uint4*)&Ksm[r][c] = *(const uint4*)&Kh[(long long)(j0 + r) * HS_ + c];
    }
    for (int t = threadIdx.x; t < HS_ * 4; t += 256) {
      const int r = t >> 2, c = (t & 3) << 3;
      *(uint4*)&Vsm[r][c] = *(const uint4*)&Vh[(long long)r * C_DIM + j0 + c];
    }
    __syncthreads();

    // ---- S = Q K^T (two 16x16 WMMA tiles over HS=64) ----
    FAcc s[2];
#pragma unroll
    for (int nt = 0; nt < 2; nt++) {
#pragma unroll
      for (int e = 0; e < 8; e++) s[nt].f[e] = 0.f;
      BFrag k0f, k1f;
      const int r = nt * 16 + lrow;
      k0f.q[0] = *(const uint4*)&Ksm[r][lhalf * 8];
      k0f.q[1] = *(const uint4*)&Ksm[r][lhalf * 8 + 16];
      k1f.q[0] = *(const uint4*)&Ksm[r][32 + lhalf * 8];
      k1f.q[1] = *(const uint4*)&Ksm[r][48 + lhalf * 8];
      s[nt].v = wmma_bf16(q0, k0f, s[nt].v);
      s[nt].v = wmma_bf16(q1, k1f, s[nt].v);
    }

    // ---- scale (+ causal mask only on diagonal blocks), online softmax ----
    float rmax[8];
    if (j0 + 31 > m0) {
#pragma unroll
      for (int e = 0; e < 8; e++) {
        const int mg = m0 + e + 8 * lhalf;
        float v0 = s[0].f[e] * scale;
        float v1 = s[1].f[e] * scale;
        if (j0 + lrow > mg)      v0 = -1e30f;
        if (j0 + 16 + lrow > mg) v1 = -1e30f;
        s[0].f[e] = v0; s[1].f[e] = v1;
        rmax[e] = fmaxf(v0, v1);
      }
    } else {
#pragma unroll
      for (int e = 0; e < 8; e++) {
        const float v0 = s[0].f[e] * scale;
        const float v1 = s[1].f[e] * scale;
        s[0].f[e] = v0; s[1].f[e] = v1;
        rmax[e] = fmaxf(v0, v1);
      }
    }
#pragma unroll
    for (int msk = 1; msk < 16; msk <<= 1)
#pragma unroll
      for (int e = 0; e < 8; e++)
        rmax[e] = fmaxf(rmax[e], __shfl_xor(rmax[e], msk, 32));

    float alpha[8], rsum[8];
#pragma unroll
    for (int e = 0; e < 8; e++) {
      const float mn = fmaxf(mrow[e], rmax[e]);
      alpha[e] = __expf(mrow[e] - mn);
      mrow[e]  = mn;
      const float p0 = __expf(s[0].f[e] - mn);
      const float p1 = __expf(s[1].f[e] - mn);
      s[0].f[e] = p0; s[1].f[e] = p1;
      rsum[e] = p0 + p1;
    }
#pragma unroll
    for (int msk = 1; msk < 16; msk <<= 1)
#pragma unroll
      for (int e = 0; e < 8; e++)
        rsum[e] += __shfl_xor(rsum[e], msk, 32);
#pragma unroll
    for (int e = 0; e < 8; e++) lsum[e] = lsum[e] * alpha[e] + rsum[e];
#pragma unroll
    for (int t = 0; t < 4; t++)
#pragma unroll
      for (int e = 0; e < 8; e++) o[t].f[e] *= alpha[e];

    // ---- relayout P (D-layout -> A-fragment layout) via per-wave LDS ----
#pragma unroll
    for (int nt = 0; nt < 2; nt++)
#pragma unroll
      for (int e = 0; e < 8; e++)
        Psm[wave][e + 8 * lhalf][nt * 16 + lrow] = f32_to_bf16(s[nt].f[e]);
    __syncthreads();

    BFrag pf;
    pf.q[0] = *(const uint4*)&Psm[wave][lrow][lhalf * 8];
    pf.q[1] = *(const uint4*)&Psm[wave][lrow][lhalf * 8 + 16];
#pragma unroll
    for (int nt = 0; nt < 4; nt++) {
      BFrag vf;
      const int r = nt * 16 + lrow;
      vf.q[0] = *(const uint4*)&Vsm[r][lhalf * 8];
      vf.q[1] = *(const uint4*)&Vsm[r][lhalf * 8 + 16];
      o[nt].v = wmma_bf16(pf, vf, o[nt].v);
    }
  }

  // ---- normalize and write concat-by-head output ----
#pragma unroll
  for (int nt = 0; nt < 4; nt++) {
#pragma unroll
    for (int e = 0; e < 8; e++) {
      const int mg  = m0 + e + 8 * lhalf;
      const int col = h * HS_ + nt * 16 + lrow;
      Obf[(long long)mg * D_DIM + col] = f32_to_bf16(o[nt].f[e] * (1.f / lsum[e]));
    }
  }
}

// ---------------------------------------------------------------------------
extern "C" void kernel_launch(void* const* d_in, const int* in_sizes, int n_in,
                              void* d_out, int out_size, void* d_ws, size_t ws_size,
                              hipStream_t stream) {
  (void)in_sizes; (void)n_in; (void)out_size; (void)ws_size;
  const float* x     = (const float*)d_in[0];
  const float* Wq    = (const float*)d_in[1];
  const float* Wk    = (const float*)d_in[2];
  const float* Wv    = (const float*)d_in[3];
  const float* Wo    = (const float*)d_in[4];
  const float* bo    = (const float*)d_in[5];
  const float* W1    = (const float*)d_in[6];
  const float* b1    = (const float*)d_in[7];
  const float* W2    = (const float*)d_in[8];
  const float* b2    = (const float*)d_in[9];
  const float* ln1_g = (const float*)d_in[10];
  const float* ln1_b = (const float*)d_in[11];
  const float* ln2_g = (const float*)d_in[12];
  const float* ln2_b = (const float*)d_in[13];
  float* out = (float*)d_out;

  char* p = (char*)d_ws;
  auto alloc = [&](size_t bytes) -> void* {
    void* r = (void*)p;
    p += (bytes + 255) & ~(size_t)255;
    return r;
  };
  unsigned short* hbf  = (unsigned short*)alloc((size_t)C_DIM * E_DIM * 2);
  unsigned short* WqT  = (unsigned short*)alloc((size_t)NH_ * HS_ * E_DIM * 2);
  unsigned short* WkT  = (unsigned short*)alloc((size_t)NH_ * HS_ * E_DIM * 2);
  unsigned short* WvT  = (unsigned short*)alloc((size_t)NH_ * HS_ * E_DIM * 2);
  unsigned short* WoT  = (unsigned short*)alloc((size_t)D_DIM * D_DIM * 2);
  unsigned short* W1T  = (unsigned short*)alloc((size_t)FF_DIM * D_DIM * 2);
  unsigned short* W2T  = (unsigned short*)alloc((size_t)E_DIM * FF_DIM * 2);
  unsigned short* Qb   = (unsigned short*)alloc((size_t)NH_ * C_DIM * HS_ * 2);
  unsigned short* Kb   = (unsigned short*)alloc((size_t)NH_ * C_DIM * HS_ * 2);
  unsigned short* Vtb  = (unsigned short*)alloc((size_t)NH_ * HS_ * C_DIM * 2);
  unsigned short* attn = (unsigned short*)alloc((size_t)C_DIM * D_DIM * 2);
  float*          x1   = (float*)alloc((size_t)C_DIM * E_DIM * 4);
  unsigned short* h2   = (unsigned short*)alloc((size_t)C_DIM * D_DIM * 2);
  unsigned short* ff1  = (unsigned short*)alloc((size_t)C_DIM * FF_DIM * 2);

  const dim3 blk(256);

  // ---- weight conversion + transpose to N-major bf16 ----
  cvt_transpose_kernel<<<dim3(HS_ / 32, E_DIM / 32, NH_), blk, 0, stream>>>(Wq, WqT, E_DIM, HS_);
  cvt_transpose_kernel<<<dim3(HS_ / 32, E_DIM / 32, NH_), blk, 0, stream>>>(Wk, WkT, E_DIM, HS_);
  cvt_transpose_kernel<<<dim3(HS_ / 32, E_DIM / 32, NH_), blk, 0, stream>>>(Wv, WvT, E_DIM, HS_);
  cvt_transpose_kernel<<<dim3(D_DIM / 32, D_DIM / 32, 1), blk, 0, stream>>>(Wo, WoT, D_DIM, D_DIM);
  cvt_transpose_kernel<<<dim3(FF_DIM / 32, D_DIM / 32, 1), blk, 0, stream>>>(W1, W1T, D_DIM, FF_DIM);
  cvt_transpose_kernel<<<dim3(E_DIM / 32, FF_DIM / 32, 1), blk, 0, stream>>>(W2, W2T, FF_DIM, E_DIM);

  // ---- LN1 -> bf16 ----
  ln_bf16_kernel<<<dim3(C_DIM), blk, 0, stream>>>(x, ln1_g, ln1_b, hbf);

  // ---- QKV projections (batched over heads); V written transposed ----
  gemm_bf16_kernel<128, 64, false, false, false, true, false>
      <<<dim3(1, C_DIM / 128, NH_), blk, 0, stream>>>(
          hbf, WqT, nullptr, nullptr, nullptr, Qb, C_DIM, HS_, E_DIM,
          0LL, (long long)HS_ * E_DIM, (long long)C_DIM * HS_);
  gemm_bf16_kernel<128, 64, false, false, false, true, false>
      <<<dim3(1, C_DIM / 128, NH_), blk, 0, stream>>>(
          hbf, WkT, nullptr, nullptr, nullptr, Kb, C_DIM, HS_, E_DIM,
          0LL, (long long)HS_ * E_DIM, (long long)C_DIM * HS_);
  gemm_bf16_kernel<128, 64, false, false, false, true, true>
      <<<dim3(1, C_DIM / 128, NH_), blk, 0, stream>>>(
          hbf, WvT, nullptr, nullptr, nullptr, Vtb, C_DIM, HS_, E_DIM,
          0LL, (long long)HS_ * E_DIM, (long long)HS_ * C_DIM);

  // ---- causal flash attention (scale = 1/sqrt(E) per reference) ----
  flash_attn_kernel<<<dim3(C_DIM / 128, NH_), blk, 0, stream>>>(
      Qb, Kb, Vtb, attn, 0.03125f);

  // ---- output projection + bias + residual -> x1 (f32) ----
  gemm_bf16_kernel<128, 128, true, false, true, false, false>
      <<<dim3(D_DIM / 128, C_DIM / 128, 1), blk, 0, stream>>>(
          attn, WoT, bo, x, x1, nullptr, C_DIM, D_DIM, D_DIM, 0LL, 0LL, 0LL);

  // ---- LN2 -> bf16 ----
  ln_bf16_kernel<<<dim3(C_DIM), blk, 0, stream>>>(x1, ln2_g, ln2_b, h2);

  // ---- FFN1: relu(h2 @ W1 + b1) -> bf16 ----
  gemm_bf16_kernel<128, 128, true, true, false, true, false>
      <<<dim3(FF_DIM / 128, C_DIM / 128, 1), blk, 0, stream>>>(
          h2, W1T, b1, nullptr, nullptr, ff1, C_DIM, FF_DIM, D_DIM, 0LL, 0LL, 0LL);

  // ---- FFN2: ff1 @ W2 + b2 + x1 -> d_out (f32) ----
  gemm_bf16_kernel<128, 128, true, false, true, false, false>
      <<<dim3(E_DIM / 128, C_DIM / 128, 1), blk, 0, stream>>>(
          ff1, W2T, b2, x1, out, nullptr, C_DIM, E_DIM, FF_DIM, 0LL, 0LL, 0LL);
}
